// MultiHeadAttention_18047452578073
// MI455X (gfx1250) — matrix operand order repaired and checked by reading, not compiled, not run on hardware
//
#include <hip/hip_runtime.h>

// MI455X / gfx1250, wave32. bf16 WMMA (16x16x32, f32 accum) for all GEMMs.
// ALL matrix operands (inputs, weights, intermediates) are converted once to
// WMMA fragment order in scratch, so every GEMM inner loop is purely
// global_load_b128 -> v_wmma.  Attention loops run two independent WMMA
// accumulator chains to hide the bf16 WMMA->WMMA hazard (no v_nop bubbles).

typedef __attribute__((ext_vector_type(16))) __bf16 v16bf;
typedef __attribute__((ext_vector_type(4)))  __bf16 v4bf;
typedef __attribute__((ext_vector_type(8)))  float  v8f;

#define BDIM 2
#define SDIM 2048
#define DDIM 1024
#define HDIM 16
#define DEPTH 64
#define NEGV (-1.0e9f)

__device__ __forceinline__ v8f wmma_bf16(v16bf a, v16bf b, v8f c) {
  // D = A(16x32 bf16) * B(32x16 bf16) + C(16x16 f32)
  return __builtin_amdgcn_wmma_f32_16x16x32_bf16(false, a, false, b, (short)0, c,
                                                 false, false);
}

// A-fragment K offset for element i (0..15), half-wave g (0/1).  Contiguous k = 16g + i (this file's own order).
__device__ __forceinline__ int a_koff(int i, int g) {
  return (g << 4) + i;
}

// Fragment-order tensor: one fragment = 32 lanes x 16 bf16 = 1KB, lane-major.
// Load = 32 contiguous bytes per lane -> two global_load_b128, coalesced.
__device__ __forceinline__ v16bf load_frag(const __bf16* __restrict__ fbase) {
  return *(const v16bf*)(fbase + ((threadIdx.x & 31) << 4));
}

// ---- fragment-order indices ----
// Generic A-operand (row-major [M x 1024] source): frag(m/16, k/32)
__device__ __forceinline__ long a_idx(int m, int k) {
  const long frag = (long)(m >> 4) * 32 + (k >> 5);
  const int lane = (m & 15) + (((k >> 4) & 1) << 4);
  return frag * 512 + lane * 16 + (k & 15);
}
// Generic B-operand (weights, [1024 k x 1024 n]): frag(k/32, n/16)
__device__ __forceinline__ long w_idx(int k, int n) {
  const long frag = (long)(k >> 5) * 64 + (n >> 4);
  const int lane = (n & 15) + (((k >> 4) & 1) << 4);
  return frag * 512 + lane * 16 + (k & 15);
}
// Qh as QK^T A-operand: frag(bh, s/16, d/32)
__device__ __forceinline__ long q_idx(int m, int n) {
  const int s = m & 2047, h = n >> 6, d = n & 63;
  const long bh = (long)(m >> 11) * HDIM + h;
  const long frag = (bh * 128 + (s >> 4)) * 2 + (d >> 5);
  const int lane = (s & 15) + (((d >> 4) & 1) << 4);
  return frag * 512 + lane * 16 + (d & 15);
}
// Kh as QK^T B-operand: frag(bh, s/16, d/32); K-dim = d
__device__ __forceinline__ long k_idx(int m, int n) {
  const int s = m & 2047, h = n >> 6, d = n & 63;
  const long bh = (long)(m >> 11) * HDIM + h;
  const long frag = (bh * 128 + (s >> 4)) * 2 + (d >> 5);
  const int lane = (s & 15) + (((d >> 4) & 1) << 4);
  return frag * 512 + lane * 16 + (d & 15);
}
// Vh as AV B-operand: frag(bh, s/32, d/16); K-dim = s, N = d
__device__ __forceinline__ long v_idx(int m, int n) {
  const int s = m & 2047, h = n >> 6, d = n & 63;
  const long bh = (long)(m >> 11) * HDIM + h;
  const long frag = (bh * 64 + (s >> 5)) * 4 + (d >> 4);
  const int lane = (d & 15) + (((s >> 4) & 1) << 4);
  return frag * 512 + lane * 16 + (s & 15);
}

// ---------------------------------------------------------------------------
// One-shot layout conversions f32 -> fragment-order bf16 (coalesced float4
// reads, 8-byte packed stores). Negligible traffic vs 23.3 TB/s HBM.
// ---------------------------------------------------------------------------
__global__ void convert_a_kernel(const float* __restrict__ X,
                                 __bf16* __restrict__ dst) {
  const int t = blockIdx.x * 256 + threadIdx.x;  // rows*256 threads
  const int m = t >> 8;
  const int k = (t & 255) << 2;
  const float4 v = *(const float4*)&X[(long)m * DDIM + k];
  v4bf p;
  p[0] = (__bf16)v.x; p[1] = (__bf16)v.y; p[2] = (__bf16)v.z; p[3] = (__bf16)v.w;
  *(v4bf*)&dst[a_idx(m, k)] = p;  // quad stays in one lane slot, 8B aligned
}
__global__ void convert_b_kernel(const float* __restrict__ W,
                                 __bf16* __restrict__ dst) {
  const int t = blockIdx.x * 256 + threadIdx.x;  // 1024*256 threads
  const int n = t & 1023;
  const int k = (t >> 10) << 2;
  v4bf p;
#pragma unroll
  for (int j = 0; j < 4; ++j) p[j] = (__bf16)W[(long)(k + j) * DDIM + n];
  *(v4bf*)&dst[w_idx(k, n)] = p;  // quad stays in one lane slot, 8B aligned
}

// ---------------------------------------------------------------------------
// Projection: dst = fragment_order(head_split(Xfrag @ Wfrag + bias)) in bf16.
// Block 256 (8 waves) -> 128x64 tile; wave -> 32x32 (2x2 subtiles).
// Inner loop: 4 fragment loads (8x global_load_b128) -> 4 independent WMMAs.
// MODE 0: q_idx layout, 1: k_idx, 2: v_idx.
// ---------------------------------------------------------------------------
template <int MODE>
__global__ void proj_kernel(const __bf16* __restrict__ xfrag,
                            const __bf16* __restrict__ wfrag,
                            const float* __restrict__ bias,
                            __bf16* __restrict__ dst) {
  const int row0 = blockIdx.x * 128;
  const int col0 = blockIdx.y * 64;
  const int wid = threadIdx.x >> 5;
  const int l = threadIdx.x & 31;
  const int g = l >> 4, nl = l & 15;
  const long r0 = row0 + (wid >> 1) * 32;
  const long c0 = col0 + (wid & 1) * 32;
  const long mt0 = r0 >> 4;
  const long nt0 = c0 >> 4;

  v8f acc00 = {}, acc01 = {}, acc10 = {}, acc11 = {};
  for (int k = 0; k < DDIM; k += 32) {
    const long kt = k >> 5;
    v16bf a0 = load_frag(xfrag + (mt0 * 32 + kt) * 512);
    v16bf a1 = load_frag(xfrag + ((mt0 + 1) * 32 + kt) * 512);
    v16bf b0 = load_frag(wfrag + (kt * 64 + nt0) * 512);
    v16bf b1 = load_frag(wfrag + (kt * 64 + nt0 + 1) * 512);
    acc00 = wmma_bf16(a0, b0, acc00);
    acc01 = wmma_bf16(a0, b1, acc01);
    acc10 = wmma_bf16(a1, b0, acc10);
    acc11 = wmma_bf16(a1, b1, acc11);
  }
#pragma unroll
  for (int rt = 0; rt < 2; ++rt) {
#pragma unroll
    for (int ct = 0; ct < 2; ++ct) {
      v8f c = rt ? (ct ? acc11 : acc10) : (ct ? acc01 : acc00);
      const int n = (int)c0 + ct * 16 + nl;
      const float bv = bias[n];
#pragma unroll
      for (int r = 0; r < 8; ++r) {
        const int m = (int)r0 + rt * 16 + r + 8 * g;
        const float v = c[r] + bv;
        long idx;
        if (MODE == 0) idx = q_idx(m, n);
        else if (MODE == 1) idx = k_idx(m, n);
        else idx = v_idx(m, n);
        dst[idx] = (__bf16)v;
      }
    }
  }
}

// ---------------------------------------------------------------------------
// Fused attention for one (b, h, 16-query-row) chunk.
// P = exp(Qh Kh^T * scale + mask*NEG) in LDS [16][2048] (131 KB of 320 KB WGP
// LDS); row sums via ds_add_f32; AV straight from LDS; attn streamed to HBM
// once (coalesced float4).  Two independent WMMA chains per loop.
// ---------------------------------------------------------------------------
__global__ void attn_kernel(const __bf16* __restrict__ qfrag,
                            const __bf16* __restrict__ kfrag,
                            const __bf16* __restrict__ vfrag,
                            const float* __restrict__ mask,
                            float* __restrict__ attn_out,
                            __bf16* __restrict__ out_heads) {
  extern __shared__ float smem[];
  float* P = smem;                 // 16*2048
  float* rowsum = P + 16 * SDIM;   // 16
  float* invs = rowsum + 16;       // 16
  float* outred = invs + 16;       // 4*16*16

  const int qt = blockIdx.x;  // 0..127
  const int h = blockIdx.y;   // 0..15
  const int b = blockIdx.z;   // 0..1
  const int wid = threadIdx.x >> 5;
  const int l = threadIdx.x & 31;
  const int g = l >> 4, nl = l & 15;
  const long bh = (long)b * HDIM + h;
  const long q0 = (long)qt * 16;
  const float scale = 0.125f;  // 1/sqrt(64)

  if (threadIdx.x < 16) rowsum[threadIdx.x] = 0.0f;
  __syncthreads();

  // Q fragments (two depth halves), loop-invariant.
  const __bf16* qf = qfrag + (bh * 128 + qt) * 2 * 512;
  v16bf aq0 = load_frag(qf);
  v16bf aq1 = load_frag(qf + 512);

  float rs[8];
#pragma unroll
  for (int r = 0; r < 8; ++r) rs[r] = 0.0f;

  // Two K-tiles per iteration (kt, kt+64): independent WMMA chains cA/cB.
  for (int kt = wid; kt < 64; kt += 8) {
    const __bf16* kfA = kfrag + (bh * 128 + kt) * 2 * 512;
    const __bf16* kfB = kfrag + (bh * 128 + kt + 64) * 2 * 512;
    __builtin_prefetch(kfA + 8 * 2 * 512 + ((threadIdx.x & 31) << 4), 0, 1);
    __builtin_prefetch(kfB + 8 * 2 * 512 + ((threadIdx.x & 31) << 4), 0, 1);
    v16bf b0A = load_frag(kfA);
    v16bf b1A = load_frag(kfA + 512);
    v16bf b0B = load_frag(kfB);
    v16bf b1B = load_frag(kfB + 512);
    v8f cA = {}, cB = {};
    cA = wmma_bf16(aq0, b0A, cA);
    cB = wmma_bf16(aq0, b0B, cB);
    cA = wmma_bf16(aq1, b1A, cA);
    cB = wmma_bf16(aq1, b1B, cB);
    const int kcolA = kt * 16 + nl;
    const int kcolB = kcolA + 64 * 16;
    const float mvA = mask[b * SDIM + kcolA] * NEGV;
    const float mvB = mask[b * SDIM + kcolB] * NEGV;
#pragma unroll
    for (int r = 0; r < 8; ++r) {
      const float pA = __expf(cA[r] * scale + mvA);  // masked -> 0
      const float pB = __expf(cB[r] * scale + mvB);
      rs[r] += pA + pB;
      P[(r + 8 * g) * SDIM + kcolA] = pA;
      P[(r + 8 * g) * SDIM + kcolB] = pB;
    }
  }
#pragma unroll
  for (int r = 0; r < 8; ++r) atomicAdd(&rowsum[r + 8 * g], rs[r]);
  __syncthreads();
  if (threadIdx.x < 16) invs[threadIdx.x] = 1.0f / rowsum[threadIdx.x];
  __syncthreads();

  // AV: out[16][64] = P @ Vh; wave -> (d-tile nt, K-half kh2).
  // Two 32-wide K-chunks per iteration: independent chains co0/co1.
  const int nt = wid & 3;
  const int kh2 = wid >> 2;
  v8f co0 = {}, co1 = {};
  for (int kk = kh2 * 1024; kk < (kh2 + 1) * 1024; kk += 64) {
    const __bf16* vf = vfrag + ((bh * 64 + (kk >> 5)) * 4 + nt) * 512;
    __builtin_prefetch(vf + 8 * 512 + ((threadIdx.x & 31) << 4), 0, 1);
    v16bf a0, a1;
#pragma unroll
    for (int i = 0; i < 16; ++i) {
      a0[i] = (__bf16)P[nl * SDIM + kk + a_koff(i, g)];       // ds b128s
      a1[i] = (__bf16)P[nl * SDIM + kk + 32 + a_koff(i, g)];
    }
    v16bf bv0 = load_frag(vf);
    v16bf bv1 = load_frag(vf + 4 * 512);
    co0 = wmma_bf16(a0, bv0, co0);
    co1 = wmma_bf16(a1, bv1, co1);
  }
  v8f co;
#pragma unroll
  for (int r = 0; r < 8; ++r) co[r] = co0[r] + co1[r];

  if (kh2 == 1) {
#pragma unroll
    for (int r = 0; r < 8; ++r)
      outred[(nt * 16 + r + 8 * g) * 16 + nl] = co[r];
  }
  __syncthreads();
  if (kh2 == 0) {
#pragma unroll
    for (int r = 0; r < 8; ++r) {
      const int m = r + 8 * g;
      const float v = (co[r] + outred[(nt * 16 + m) * 16 + nl]) * invs[m];
      const int mrow = b * SDIM + (int)q0 + m;          // row in [0,4096)
      const int ncol = h * DEPTH + nt * 16 + nl;        // col in [0,1024)
      out_heads[a_idx(mrow, ncol)] = (__bf16)v;         // A-frag order for Wo
    }
  }

  // Stream normalized attn tile to HBM once, coalesced float4.
  float* abase = attn_out + (bh * SDIM + q0) * SDIM;
#pragma unroll 1
  for (int it = 0; it < 32; ++it) {
    const int e = it * 1024 + (int)threadIdx.x * 4;  // linear over [16][2048]
    const int row = e >> 11, col = e & 2047;
    const float inv = invs[row];
    float4 pv = *(const float4*)&P[e];
    float4 o = make_float4(pv.x * inv, pv.y * inv, pv.z * inv, pv.w * inv);
    *(float4*)&abase[(long)row * SDIM + col] = o;
  }
}

// ---------------------------------------------------------------------------
// out = out_heads(frag bf16) @ Wo(frag bf16) + bo -> f32 [4096,1024]
// ---------------------------------------------------------------------------
__global__ void out_proj_kernel(const __bf16* __restrict__ ohfrag,
                                const __bf16* __restrict__ wfrag,
                                const float* __restrict__ bias,
                                float* __restrict__ out) {
  const int row0 = blockIdx.x * 128;
  const int col0 = blockIdx.y * 64;
  const int wid = threadIdx.x >> 5;
  const int l = threadIdx.x & 31;
  const int g = l >> 4, nl = l & 15;
  const long r0 = row0 + (wid >> 1) * 32;
  const long c0 = col0 + (wid & 1) * 32;
  const long mt0 = r0 >> 4;
  const long nt0 = c0 >> 4;

  v8f acc00 = {}, acc01 = {}, acc10 = {}, acc11 = {};
  for (int k = 0; k < DDIM; k += 32) {
    const long kt = k >> 5;
    v16bf a0 = load_frag(ohfrag + (mt0 * 32 + kt) * 512);
    v16bf a1 = load_frag(ohfrag + ((mt0 + 1) * 32 + kt) * 512);
    v16bf b0 = load_frag(wfrag + (kt * 64 + nt0) * 512);
    v16bf b1 = load_frag(wfrag + (kt * 64 + nt0 + 1) * 512);
    acc00 = wmma_bf16(a0, b0, acc00);
    acc01 = wmma_bf16(a0, b1, acc01);
    acc10 = wmma_bf16(a1, b0, acc10);
    acc11 = wmma_bf16(a1, b1, acc11);
  }
#pragma unroll
  for (int rt = 0; rt < 2; ++rt) {
#pragma unroll
    for (int ct = 0; ct < 2; ++ct) {
      v8f c = rt ? (ct ? acc11 : acc10) : (ct ? acc01 : acc00);
      const int n = (int)c0 + ct * 16 + nl;
      const float bv = bias[n];
#pragma unroll
      for (int r = 0; r < 8; ++r) {
        const long m = r0 + rt * 16 + r + 8 * g;
        out[m * DDIM + n] = c[r] + bv;
      }
    }
  }
}

extern "C" void kernel_launch(void* const* d_in, const int* in_sizes, int n_in,
                              void* d_out, int out_size, void* d_ws,
                              size_t ws_size, hipStream_t stream) {
  const float* q = (const float*)d_in[0];
  const float* k = (const float*)d_in[1];
  const float* v = (const float*)d_in[2];
  const float* mask = (const float*)d_in[3];
  const float* wq = (const float*)d_in[4];
  const float* bq = (const float*)d_in[5];
  const float* wk = (const float*)d_in[6];
  const float* bk = (const float*)d_in[7];
  const float* wv = (const float*)d_in[8];
  const float* bv = (const float*)d_in[9];
  const float* wo = (const float*)d_in[10];
  const float* bo = (const float*)d_in[11];

  const size_t n_act = (size_t)BDIM * SDIM * DDIM;  // 4,194,304 elements
  char* ws = (char*)d_ws;
  __bf16* qfrag = (__bf16*)ws;                    // 8 MB
  __bf16* kfrag = (__bf16*)(ws + 2 * n_act);      // 8 MB
  __bf16* vfrag = (__bf16*)(ws + 4 * n_act);      // 8 MB
  __bf16* ohfrag = (__bf16*)(ws + 6 * n_act);     // 8 MB
  __bf16* xfrag = (__bf16*)(ws + 8 * n_act);      // 8 MB (reused q/k/v)
  __bf16* wfrag = (__bf16*)(ws + 10 * n_act);     // 2 MB (reused per weight)
  float* out = (float*)d_out;                     // [B,S,D]
  float* attn = out + n_act;                      // [B,H,S,S]

  dim3 blk(256);
  dim3 g1(4096 / 128, DDIM / 64);          // (32, 16) GEMM grid
  const int gca = 4096;                    // convert_a: 4096 rows * 256 thr
  const int gcb = 1024;                    // convert_b: 1024*1024/4 elems

  // Q path
  convert_a_kernel<<<gca, blk, 0, stream>>>(q, xfrag);
  convert_b_kernel<<<gcb, blk, 0, stream>>>(wq, wfrag);
  proj_kernel<0><<<g1, blk, 0, stream>>>(xfrag, wfrag, bq, qfrag);
  // K path
  convert_a_kernel<<<gca, blk, 0, stream>>>(k, xfrag);
  convert_b_kernel<<<gcb, blk, 0, stream>>>(wk, wfrag);
  proj_kernel<1><<<g1, blk, 0, stream>>>(xfrag, wfrag, bk, kfrag);
  // V path
  convert_a_kernel<<<gca, blk, 0, stream>>>(v, xfrag);
  convert_b_kernel<<<gcb, blk, 0, stream>>>(wv, wfrag);
  proj_kernel<2><<<g1, blk, 0, stream>>>(xfrag, wfrag, bv, vfrag);

  dim3 g2(SDIM / 16, HDIM, BDIM);  // (128, 16, 2)
  const size_t lds = (size_t)(16 * SDIM + 16 + 16 + 4 * 16 * 16) * sizeof(float);
  attn_kernel<<<g2, blk, lds, stream>>>(qfrag, kfrag, vfrag, mask, attn, ohfrag);

  convert_b_kernel<<<gcb, blk, 0, stream>>>(wo, wfrag);
  out_proj_kernel<<<g1, blk, 0, stream>>>(ohfrag, wfrag, bo, out);
}
